// SpecializedGPT_43997644980950
// MI455X (gfx1250) — matrix operand order repaired and checked by reading, not compiled, not run on hardware
//
#include <hip/hip_runtime.h>
#include <hip/hip_bf16.h>

// ---------------------------------------------------------------------------
// SpecializedGPT forward on gfx1250 (MI455X)
//  - all GEMMs via v_wmma_f32_16x16x32_f16 (f16 in, f32 accumulate)
//  - weights pre-transposed to [N][K] f16 so B-frags are contiguous b128 loads
//  - B/K tiles staged to LDS with GLOBAL_LOAD_ASYNC_TO_LDS_B128 (ASYNCcnt)
//  - attention: flash-style, K + V^T staged in LDS, all frag loads b128,
//    softmax cross-lane reductions in pure VALU via DPP8 / DPP16 row_ror:8
// ---------------------------------------------------------------------------

typedef _Float16 h16_t;
typedef __attribute__((ext_vector_type(16))) _Float16 v16h;
typedef __attribute__((ext_vector_type(8)))  _Float16 v8h;
typedef __attribute__((ext_vector_type(8)))  float    v8f;

#define NB   128
#define NT   256
#define ND   256
#define NH   8
#define NL   4
#define NHS  32
#define NM   (NB * NT)        // 32768 flattened tokens
#define NVP  128
#define NVR  64
#define NFF  1024             // 4*D
#define LDSB 40               // padded LDS row stride (halves): 80B, 16B-aligned,
                              // 20*n mod 64 banks -> conflict-free b128 reads
#define VTS  264              // V^T row stride (halves): 528B, 4*n mod 64 banks

// DPP8 selectors: lane i reads lane sel[i] within each group of 8
#define DPP8_XOR1 ((1) | (0 << 3) | (3 << 6) | (2 << 9) | (5 << 12) | (4 << 15) | (7 << 18) | (6 << 21))
#define DPP8_XOR2 ((2) | (3 << 3) | (0 << 6) | (1 << 9) | (6 << 12) | (7 << 15) | (4 << 18) | (5 << 21))
#define DPP8_XOR4 ((4) | (5 << 3) | (6 << 6) | (7 << 9) | (0 << 12) | (1 << 15) | (2 << 18) | (3 << 21))
#define DPP16_ROW_ROR8 0x128

static __device__ __forceinline__ v8f vzero() {
    v8f z;
#pragma unroll
    for (int i = 0; i < 8; ++i) z[i] = 0.0f;
    return z;
}

static __device__ __forceinline__ v8f wmma_f16(v16h a, v16h b, v8f c) {
    return __builtin_amdgcn_wmma_f32_16x16x32_f16(
        false, a, false, b, (short)0, c, false, false);
}

static __device__ __forceinline__ v16h frag_from2(v8h lo, v8h hi) {
    v16h a;
#pragma unroll
    for (int j = 0; j < 8; ++j) { a[j] = lo[j]; a[8 + j] = hi[j]; }
    return a;
}

// A fragment: 16x32 f16 row-major. Lane m=lane&15, kb=8*(lane>=16):
// slots 0..7 = K kb..kb+7 (16B run), slots 8..15 = K 16+kb..16+kb+7 (16B run).
static __device__ __forceinline__ v16h load_frag_a(const h16_t* __restrict__ p, int ld) {
    const int lane = threadIdx.x & 31;
    const h16_t* r = p + (size_t)(lane & 15) * ld + ((lane >> 4) * 8);
    return frag_from2(*(const v8h*)r, *(const v8h*)(r + 16));
}

// B fragment from row-major [N=16, K=32] source:
// lane n=lane&15, kb=8*(lane>=16): slots 0..7 = K kb..kb+7, slots 8..15 = K 16+kb..16+kb+7.
static __device__ __forceinline__ v16h load_frag_b_nk(const h16_t* __restrict__ p, int ld) {
    const int lane = threadIdx.x & 31;
    const h16_t* r = p + (size_t)(lane & 15) * ld + ((lane >> 4) * 8);
    return frag_from2(*(const v8h*)r, *(const v8h*)(r + 16));
}

// 16-lane (half-wave) reductions in pure VALU: DPP8 xor1/2/4 + DPP16 row_ror:8.
static __device__ __forceinline__ float redmax16(float x) {
    x = fmaxf(x, __int_as_float(__builtin_amdgcn_mov_dpp8(__float_as_int(x), DPP8_XOR1)));
    x = fmaxf(x, __int_as_float(__builtin_amdgcn_mov_dpp8(__float_as_int(x), DPP8_XOR2)));
    x = fmaxf(x, __int_as_float(__builtin_amdgcn_mov_dpp8(__float_as_int(x), DPP8_XOR4)));
    x = fmaxf(x, __int_as_float(__builtin_amdgcn_update_dpp(
            __float_as_int(x), __float_as_int(x), DPP16_ROW_ROR8, 0xf, 0xf, true)));
    return x;
}
static __device__ __forceinline__ float redsum16(float x) {
    x += __int_as_float(__builtin_amdgcn_mov_dpp8(__float_as_int(x), DPP8_XOR1));
    x += __int_as_float(__builtin_amdgcn_mov_dpp8(__float_as_int(x), DPP8_XOR2));
    x += __int_as_float(__builtin_amdgcn_mov_dpp8(__float_as_int(x), DPP8_XOR4));
    x += __int_as_float(__builtin_amdgcn_update_dpp(
            __float_as_int(x), __float_as_int(x), DPP16_ROW_ROR8, 0xf, 0xf, true));
    return x;
}

// CDNA5 async global->LDS copy (16B per lane), tracked by ASYNCcnt.
static __device__ __forceinline__ void async_copy_b128(const h16_t* lds_ptr,
                                                       const h16_t* gptr) {
    const uint32_t lds_off = (uint32_t)(uintptr_t)lds_ptr;  // low 32b = LDS offset
    asm volatile("global_load_async_to_lds_b128 %0, %1, off"
                 :: "v"(lds_off), "v"(gptr) : "memory");
}
static __device__ __forceinline__ void wait_async0() {
    asm volatile("s_wait_asynccnt 0" ::: "memory");
}
static __device__ __forceinline__ void wait_async1() {
    asm volatile("s_wait_asynccnt 1" ::: "memory");
}

// ---------------------------------------------------------------------------
// GEMM: C[M,N] = act( A[M,K](f16) * W^T[N,K](f16) + bias[N] (+ R[M,N]) )
// 256 threads = 8 waves; block tile 128(M) x 64(N); wave tile 16 x 64.
// W^T tile (64 rows x 32 K) async-staged to LDS, double-buffered.
// ---------------------------------------------------------------------------
template <bool RELU, bool RESID>
__global__ __launch_bounds__(256) void gemm_wmma_kernel(
    const h16_t* __restrict__ A, const h16_t* __restrict__ BwT,
    const float* __restrict__ bias, const float* __restrict__ R,
    float* __restrict__ Cf, h16_t* __restrict__ C16,
    int M, int N, int K, int lda, int ldc) {
    __shared__ __align__(16) h16_t sBT[2][64 * LDSB];

    const int lane = threadIdx.x & 31;
    const int wave = threadIdx.x >> 5;
    const int m0   = blockIdx.y * 128 + wave * 16;
    const int n0   = blockIdx.x * 64;

    v8f acc[4];
#pragma unroll
    for (int j = 0; j < 4; ++j) acc[j] = vzero();

    // staging: thread t copies 16B: W^T row n0+(t>>2), K halves k0+(t&3)*8
    const int cr = threadIdx.x >> 2;
    const int cs = (threadIdx.x & 3) * 8;
    const h16_t* gsrc = BwT + (size_t)(n0 + cr) * K + cs;

    async_copy_b128(&sBT[0][cr * LDSB + cs], gsrc);

    for (int k0 = 0; k0 < K; k0 += 32) {
        const int buf = (k0 >> 5) & 1;
        if (k0 + 32 < K) {
            async_copy_b128(&sBT[buf ^ 1][cr * LDSB + cs], gsrc + k0 + 32);
            wait_async1();                 // oldest copy (this tile) done
            __builtin_prefetch(A + (size_t)m0 * lda + k0 + 32, 0, 0);
        } else {
            wait_async0();
        }
        __syncthreads();

        v16h a = load_frag_a(A + (size_t)m0 * lda + k0, lda);
#pragma unroll
        for (int j = 0; j < 4; ++j) {
            v16h b = load_frag_b_nk(&sBT[buf][j * 16 * LDSB], LDSB);
            acc[j] = wmma_f16(a, b, acc[j]);
        }
        __syncthreads();
    }

    const int cn = lane & 15;
    const int mb = (lane >> 4) * 8;
#pragma unroll
    for (int j = 0; j < 4; ++j) {
        const int col = n0 + j * 16 + cn;
        const float bv = bias ? bias[col] : 0.0f;
#pragma unroll
        for (int i = 0; i < 8; ++i) {
            const int row = m0 + mb + i;
            float v = acc[j][i] + bv;
            if (RESID) v += R[(size_t)row * ldc + col];
            if (RELU) v = fmaxf(v, 0.0f);
            if (Cf)  Cf[(size_t)row * ldc + col] = v;
            if (C16) C16[(size_t)row * ldc + col] = (h16_t)v;
        }
    }
}

// ---------------------------------------------------------------------------
// Flash-style causal attention. qkv: [NM, 768] f16 (q | k | v, 32 cols/head).
// One block per (b,h). K async-staged to LDS (padded), V staged transposed.
// 8 waves; wave w handles query tiles {w, 15-w} (balanced causal work).
// ---------------------------------------------------------------------------
__global__ __launch_bounds__(256) void attn_kernel(
    const h16_t* __restrict__ qkv, h16_t* __restrict__ o16) {
    __shared__ __align__(16) h16_t sK[NT * LDSB];     // K  [key][e], padded rows
    __shared__ __align__(16) h16_t sVT[NHS * VTS];    // V^T [e][key], padded rows
    __shared__ __align__(16) h16_t sP[8][16 * 32];    // per-wave P re-layout tile

    const int lane = threadIdx.x & 31;
    const int wave = threadIdx.x >> 5;
    const int b    = blockIdx.x / NH;
    const int h    = blockIdx.x % NH;
    const int ld   = 3 * ND;                          // 768
    const h16_t* base = qkv + (size_t)b * NT * ld;

    const int cn = lane & 15;
    const int mb = (lane >> 4) * 8;

    // ---- stage K (async b128) and V^T (coalesced read + LDS transpose) ----
    {
        const int r0  = threadIdx.x >> 2;             // 0..63
        const int seg = (threadIdx.x & 3) * 8;        // 0,8,16,24
#pragma unroll
        for (int c = 0; c < 4; ++c) {
            const int key = r0 + c * 64;
            async_copy_b128(&sK[key * LDSB + seg],
                            base + (size_t)key * ld + ND + h * NHS + seg);
        }
#pragma unroll
        for (int c = 0; c < 4; ++c) {
            const int key = r0 + c * 64;
            const v8h v = *(const v8h*)(base + (size_t)key * ld + 2 * ND + h * NHS + seg);
#pragma unroll
            for (int j = 0; j < 8; ++j) sVT[(seg + j) * VTS + key] = v[j];
        }
    }
    wait_async0();
    __syncthreads();

    for (int rep = 0; rep < 2; ++rep) {
        const int qt = rep ? (15 - wave) : wave;      // balanced pairing
        v16h qf = load_frag_a(base + (size_t)qt * 16 * ld + h * NHS, ld);
#pragma unroll
        for (int j = 0; j < 16; ++j) qf[j] *= (_Float16)0.0625f;  // D^-0.5 folded in

        v8f o0 = vzero(), o1 = vzero();
        float mrow[8], lrow[8];
#pragma unroll
        for (int i = 0; i < 8; ++i) { mrow[i] = -1e30f; lrow[i] = 0.0f; }

        const int ksteps = qt / 2 + 1;                // causal: skip masked blocks
        for (int s = 0; s < ksteps; ++s) {
            const int kb = s * 32;
            // S = Q * K^T : b128 pairs from padded LDS K tile
            v16h bk0 = load_frag_b_nk(&sK[kb * LDSB], LDSB);
            v16h bk1 = load_frag_b_nk(&sK[(kb + 16) * LDSB], LDSB);
            v8f s0 = wmma_f16(qf, bk0, vzero());
            v8f s1 = wmma_f16(qf, bk1, vzero());

#pragma unroll
            for (int i = 0; i < 8; ++i) {
                const int qrow = qt * 16 + mb + i;
                if (kb + cn > qrow)      s0[i] = -1e30f;  // causal mask
                if (kb + 16 + cn > qrow) s1[i] = -1e30f;
            }

            // online softmax; rows live in 16-lane halves -> DPP reductions
#pragma unroll
            for (int i = 0; i < 8; ++i) {
                const float mx   = redmax16(fmaxf(s0[i], s1[i]));
                const float mnew = fmaxf(mrow[i], mx);
                const float alpha = __expf(mrow[i] - mnew);
                const float p0 = __expf(s0[i] - mnew);
                const float p1 = __expf(s1[i] - mnew);
                const float rs = redsum16(p0 + p1);
                lrow[i] = lrow[i] * alpha + rs;
                mrow[i] = mnew;
                o0[i] *= alpha;
                o1[i] *= alpha;
                const int r = mb + i;
                sP[wave][r * 32 + cn]      = (h16_t)p0;
                sP[wave][r * 32 + 16 + cn] = (h16_t)p1;
            }

            // O += P * V : P re-laid C->A via per-wave LDS; V^T gives b128 frags
            v16h pa  = load_frag_a(&sP[wave][0], 32);
            v16h bv0 = load_frag_b_nk(&sVT[0 * VTS + kb], VTS);    // e cols 0..15
            v16h bv1 = load_frag_b_nk(&sVT[16 * VTS + kb], VTS);   // e cols 16..31
            o0 = wmma_f16(pa, bv0, o0);
            o1 = wmma_f16(pa, bv1, o1);
        }

#pragma unroll
        for (int i = 0; i < 8; ++i) {
            const int row = b * NT + qt * 16 + mb + i;
            const float inv = 1.0f / lrow[i];
            o16[(size_t)row * ND + h * NHS + cn]      = (h16_t)(o0[i] * inv);
            o16[(size_t)row * ND + h * NHS + 16 + cn] = (h16_t)(o1[i] * inv);
        }
    }
}

// ---------------------------------------------------------------------------
// LayerNorm over D=256; one wave per row; f16 output for WMMA consumption.
// ---------------------------------------------------------------------------
__global__ __launch_bounds__(256) void ln_kernel(
    const float* __restrict__ x, const float* __restrict__ g,
    const float* __restrict__ bb, h16_t* __restrict__ out) {
    const int lane = threadIdx.x & 31;
    const int wave = threadIdx.x >> 5;
    const int row  = blockIdx.x * 8 + wave;
    const float* xr = x + (size_t)row * ND;

    float vals[8];
    float s = 0.0f, s2 = 0.0f;
#pragma unroll
    for (int i = 0; i < 8; ++i) {
        const float v = xr[lane * 8 + i];
        vals[i] = v; s += v; s2 += v * v;
    }
#pragma unroll
    for (int d = 1; d < 32; d <<= 1) {
        s  += __shfl_xor(s, d, 32);
        s2 += __shfl_xor(s2, d, 32);
    }
    const float mean = s * (1.0f / ND);
    const float var  = s2 * (1.0f / ND) - mean * mean;
    const float rstd = rsqrtf(var + 1e-5f);
#pragma unroll
    for (int i = 0; i < 8; ++i) {
        const int e = lane * 8 + i;
        out[(size_t)row * ND + e] = (h16_t)((vals[i] - mean) * rstd * g[e] + bb[e]);
    }
}

__global__ __launch_bounds__(256) void embed_kernel(
    const int* __restrict__ pidx, const int* __restrict__ ridx,
    const float* __restrict__ pe, const float* __restrict__ re,
    const float* __restrict__ pos, float* __restrict__ x) {
    const int i = blockIdx.x * 256 + threadIdx.x;
    const int m = i / ND, d = i % ND, t = m % NT;
    x[i] = pe[(size_t)pidx[m] * ND + d] + re[(size_t)ridx[m] * ND + d] + pos[(size_t)t * ND + d];
}

// Pack per-head QKV [L,H,D,HS] f32 -> fused TRANSPOSED [L][3*D rows][D cols] f16
__global__ __launch_bounds__(256) void pack_qkv_kernel(
    const float* __restrict__ wq, const float* __restrict__ wk,
    const float* __restrict__ wv, h16_t* __restrict__ out) {
    const int i = blockIdx.x * 256 + threadIdx.x;   // over NL*NH*ND*NHS
    const int e = i % NHS;
    const int d = (i / NHS) % ND;
    const int h = (i / (NHS * ND)) % NH;
    const int l = i / (NHS * ND * NH);
    const size_t ob = (size_t)l * (3 * ND) * ND;
    const int    rowq = h * NHS + e;
    out[ob + (size_t)rowq * ND + d]              = (h16_t)wq[i];
    out[ob + (size_t)(ND + rowq) * ND + d]       = (h16_t)wk[i];
    out[ob + (size_t)(2 * ND + rowq) * ND + d]   = (h16_t)wv[i];
}

// dst[l][n*K + k] = (f16) src[l][k*N + n]   (per-layer transpose via blockIdx.y)
__global__ __launch_bounds__(256) void transpose_f16_kernel(
    const float* __restrict__ src, h16_t* __restrict__ dst, int K, int N) {
    const size_t per  = (size_t)K * N;
    const size_t base = (size_t)blockIdx.y * per;
    const int i = blockIdx.x * 256 + threadIdx.x;
    if (i >= K * N) return;
    const int k = i / N, n = i % N;
    dst[base + (size_t)n * K + k] = (h16_t)src[base + i];
}

// Per-row cross entropy: nll[row] = logsumexp(logits[row]) - logits[row][tgt]
__global__ __launch_bounds__(256) void ce_kernel(
    const float* __restrict__ logits, const int* __restrict__ tgt,
    float* __restrict__ nll, int V) {
    const int lane = threadIdx.x & 31;
    const int wave = threadIdx.x >> 5;
    const int row  = blockIdx.x * 8 + wave;
    const float* lr = logits + (size_t)row * V;
    const int per = V / 32;

    float mx = -1e30f;
    for (int i = 0; i < per; ++i) mx = fmaxf(mx, lr[lane * per + i]);
#pragma unroll
    for (int d = 1; d < 32; d <<= 1) mx = fmaxf(mx, __shfl_xor(mx, d, 32));
    float s = 0.0f;
    for (int i = 0; i < per; ++i) s += __expf(lr[lane * per + i] - mx);
#pragma unroll
    for (int d = 1; d < 32; d <<= 1) s += __shfl_xor(s, d, 32);
    if (lane == 0) nll[row] = (mx + __logf(s)) - lr[tgt[row]];
}

// Deterministic fixed-order reduction: loss = (sum_p + sum_r) / (2*NM)
__global__ __launch_bounds__(256) void loss_kernel(
    const float* __restrict__ nll, float* __restrict__ out) {
    __shared__ float sp[256], sr[256];
    float a = 0.0f, c = 0.0f;
    for (int i = threadIdx.x; i < NM; i += 256) { a += nll[i]; c += nll[NM + i]; }
    sp[threadIdx.x] = a; sr[threadIdx.x] = c;
    __syncthreads();
    for (int d = 128; d > 0; d >>= 1) {
        if ((int)threadIdx.x < d) {
            sp[threadIdx.x] += sp[threadIdx.x + d];
            sr[threadIdx.x] += sr[threadIdx.x + d];
        }
        __syncthreads();
    }
    if (threadIdx.x == 0) out[0] = (sp[0] + sr[0]) * (0.5f / NM);
}

// ---------------------------------------------------------------------------

extern "C" void kernel_launch(void* const* d_in, const int* in_sizes, int n_in,
                              void* d_out, int out_size, void* d_ws, size_t ws_size,
                              hipStream_t stream) {
    (void)in_sizes; (void)n_in; (void)out_size; (void)ws_size;

    const int*   pitch_idx  = (const int*)  d_in[0];
    const int*   rhythm_idx = (const int*)  d_in[1];
    const int*   pitch_tgt  = (const int*)  d_in[2];
    const int*   rhythm_tgt = (const int*)  d_in[3];
    const float* pitch_emb  = (const float*)d_in[4];
    const float* rhythm_emb = (const float*)d_in[5];
    const float* pos_emb    = (const float*)d_in[6];
    const float* wk         = (const float*)d_in[7];
    const float* wq         = (const float*)d_in[8];
    const float* wv         = (const float*)d_in[9];
    const float* wproj      = (const float*)d_in[10];
    const float* bproj      = (const float*)d_in[11];
    const float* ln1_g      = (const float*)d_in[12];
    const float* ln1_b      = (const float*)d_in[13];
    const float* ln2_g      = (const float*)d_in[14];
    const float* ln2_b      = (const float*)d_in[15];
    const float* w1         = (const float*)d_in[16];
    const float* b1         = (const float*)d_in[17];
    const float* w2         = (const float*)d_in[18];
    const float* b2         = (const float*)d_in[19];
    const float* lnf_g      = (const float*)d_in[20];
    const float* lnf_b      = (const float*)d_in[21];
    const float* wp_head    = (const float*)d_in[22];
    const float* bp_head    = (const float*)d_in[23];
    const float* wr_head    = (const float*)d_in[24];
    const float* br_head    = (const float*)d_in[25];

    // bump allocator over d_ws (256B aligned chunks)
    char* ws = (char*)d_ws;
    auto alloc = [&](size_t bytes) -> char* {
        char* p = ws;
        ws += (bytes + 255) & ~(size_t)255;
        return p;
    };
    float* x       = (float*)alloc((size_t)NM * ND * 4);        // residual stream f32
    h16_t* h16b    = (h16_t*)alloc((size_t)NM * ND * 2);        // LN output f16
    h16_t* o16     = (h16_t*)alloc((size_t)NM * ND * 2);        // attention output f16
    h16_t* buf16   = (h16_t*)alloc((size_t)NM * NFF * 2);       // qkv (768) / mlp act (1024)
    h16_t* wqkv16  = (h16_t*)alloc((size_t)NL * 3 * ND * ND * 2);  // [L][768][256] W^T
    h16_t* wproj16 = (h16_t*)alloc((size_t)NL * ND * ND * 2);      // [L][256][256] W^T
    h16_t* w116    = (h16_t*)alloc((size_t)NL * NFF * ND * 2);     // [L][1024][256] W^T
    h16_t* w216    = (h16_t*)alloc((size_t)NL * ND * NFF * 2);     // [L][256][1024] W^T
    h16_t* wp16    = (h16_t*)alloc((size_t)NVP * ND * 2);          // [128][256] W^T
    h16_t* wr16    = (h16_t*)alloc((size_t)NVR * ND * 2);          // [64][256] W^T
    float* nll     = (float*)alloc((size_t)2 * NM * 4);

    const dim3 blk(256);

    // --- weight f16 conversion + transpose to [N][K] (deterministic, every call) ---
    pack_qkv_kernel<<<(NL * NH * ND * NHS) / 256, blk, 0, stream>>>(wq, wk, wv, wqkv16);
    transpose_f16_kernel<<<dim3((ND * ND) / 256, NL), blk, 0, stream>>>(wproj, wproj16, ND, ND);
    transpose_f16_kernel<<<dim3((ND * NFF) / 256, NL), blk, 0, stream>>>(w1, w116, ND, NFF);
    transpose_f16_kernel<<<dim3((NFF * ND) / 256, NL), blk, 0, stream>>>(w2, w216, NFF, ND);
    transpose_f16_kernel<<<dim3((ND * NVP) / 256, 1), blk, 0, stream>>>(wp_head, wp16, ND, NVP);
    transpose_f16_kernel<<<dim3((ND * NVR) / 256, 1), blk, 0, stream>>>(wr_head, wr16, ND, NVR);

    // --- embeddings ---
    embed_kernel<<<((size_t)NM * ND) / 256, blk, 0, stream>>>(
        pitch_idx, rhythm_idx, pitch_emb, rhythm_emb, pos_emb, x);

    // --- transformer layers ---
    for (int l = 0; l < NL; ++l) {
        ln_kernel<<<NM / 8, blk, 0, stream>>>(x, ln1_g + l * ND, ln1_b + l * ND, h16b);
        // fused QKV GEMM: [M,256] x (W^T[768,256])^T -> f16 qkv
        gemm_wmma_kernel<false, false><<<dim3(768 / 64, NM / 128), blk, 0, stream>>>(
            h16b, wqkv16 + (size_t)l * 3 * ND * ND, nullptr, nullptr,
            nullptr, buf16, NM, 768, ND, ND, 768);
        attn_kernel<<<NB * NH, blk, 0, stream>>>(buf16, o16);
        // proj + bias + residual into x
        gemm_wmma_kernel<false, true><<<dim3(ND / 64, NM / 128), blk, 0, stream>>>(
            o16, wproj16 + (size_t)l * ND * ND, bproj + l * ND, x,
            x, nullptr, NM, ND, ND, ND, ND);
        ln_kernel<<<NM / 8, blk, 0, stream>>>(x, ln2_g + l * ND, ln2_b + l * ND, h16b);
        // MLP1 + b1 + ReLU -> f16 act
        gemm_wmma_kernel<true, false><<<dim3(NFF / 64, NM / 128), blk, 0, stream>>>(
            h16b, w116 + (size_t)l * NFF * ND, b1 + l * NFF, nullptr,
            nullptr, buf16, NM, NFF, ND, ND, NFF);
        // MLP2 + b2 + residual into x
        gemm_wmma_kernel<false, true><<<dim3(ND / 64, NM / 128), blk, 0, stream>>>(
            buf16, w216 + (size_t)l * ND * NFF, b2 + l * ND, x,
            x, nullptr, NM, ND, NFF, NFF, ND);
    }

    // --- final LN + heads (logits straight into d_out) ---
    ln_kernel<<<NM / 8, blk, 0, stream>>>(x, lnf_g, lnf_b, h16b);
    float* lp = (float*)d_out;
    float* lr = lp + (size_t)NM * NVP;
    gemm_wmma_kernel<false, false><<<dim3(NVP / 64, NM / 128), blk, 0, stream>>>(
        h16b, wp16, bp_head, nullptr, lp, nullptr, NM, NVP, ND, ND, NVP);
    gemm_wmma_kernel<false, false><<<dim3(NVR / 64, NM / 128), blk, 0, stream>>>(
        h16b, wr16, br_head, nullptr, lr, nullptr, NM, NVR, ND, ND, NVR);

    // --- cross entropy + deterministic loss reduction ---
    ce_kernel<<<NM / 8, blk, 0, stream>>>(lp, pitch_tgt, nll, NVP);
    ce_kernel<<<NM / 8, blk, 0, stream>>>(lr, rhythm_tgt, nll + NM, NVR);
    loss_kernel<<<1, blk, 0, stream>>>(nll, lp + (size_t)NM * (NVP + NVR));
}
